// GAT_24481313587853
// MI455X (gfx1250) — matrix operand order repaired and checked by reading, not compiled, not run on hardware
//
#include <hip/hip_runtime.h>
#include <math.h>

// ---------------- problem constants (from reference) ----------------
#define N_NODES 100000
#define N_EDGES 1600000
#define ETOT    (N_EDGES + N_NODES)   // self loops appended
#define IN_DIM  256
#define HID     128
#define HEADS   4
#define C1      32
#define OUT_C   2
#define NEG_SLOPE 0.2f

typedef __attribute__((ext_vector_type(16))) __bf16 v16bf;
typedef __attribute__((ext_vector_type(8)))  float  v8f;
typedef __attribute__((ext_vector_type(4)))  unsigned v4u;
typedef __attribute__((ext_vector_type(8)))  int      v8i;
typedef __attribute__((ext_vector_type(4)))  int      v4i;

// f32 -> bf16 round-to-nearest-even
static __device__ __forceinline__ unsigned short f2bf(float f) {
  unsigned u = __float_as_uint(f);
  unsigned lsb = (u >> 16) & 1u;
  u += 0x7FFFu + lsb;
  return (unsigned short)(u >> 16);
}
static __device__ __forceinline__ unsigned pk2(float a, float b) {
  return (unsigned)f2bf(a) | ((unsigned)f2bf(b) << 16);
}

static __device__ __forceinline__ void edge_sd(const long long* __restrict__ ei,
                                               int e, int& s, int& d) {
  if (e < N_EDGES) { s = (int)ei[e]; d = (int)ei[N_EDGES + e]; }
  else             { s = e - N_EDGES; d = s; }   // self loop
}

// ---------------- pack W1 (256x128 f32, row-major K x N) into WMMA-B bf16 layout ----
// B fragment (32x16, KxN): lane = (N & 15) + 16*khalf ; dword v holds K = (v>>2)*16+khalf*8+2*(v&3), +1
// Storage: Bp[((ktile*8 + ntile)*32 + lane)*8 + v]  (dwords), ktile in 0..7, ntile in 0..7
__global__ __launch_bounds__(256) void packW1_kernel(const float* __restrict__ W1,
                                                     unsigned* __restrict__ Bp) {
  int id = blockIdx.x * 256 + threadIdx.x;           // 0 .. 16383
  if (id >= 8 * 8 * 32 * 8) return;
  int v    =  id        & 7;
  int lane = (id >> 3)  & 31;
  int j    = (id >> 8)  & 7;
  int t    =  id >> 11;
  int n  = j * 16 + (lane & 15);
  int k0 = t * 32 + ((v >> 2) << 4) + ((lane >> 4) << 3) + 2 * (v & 3);
  Bp[id] = pk2(W1[(size_t)k0 * HID + n], W1[(size_t)(k0 + 1) * HID + n]);
}

// ---------------- GEMM1: h1raw[N x 128] = bf16(x) @ bf16(W1), f32 accum, WMMA ------
// 256-thread block = 8 waves, each wave computes a 16x128 output slab.
// Packed B (64 KB) is staged once per block into LDS via the Tensor Data Mover,
// then all 512 WMMAs of the block feed from ds_load_b128.
#define B_DWORDS (8 * 8 * 32 * 8)   // 16384 dwords = 64 KB
__global__ __launch_bounds__(256) void gemm1_wmma_kernel(const float* __restrict__ x,
                                                         const unsigned* __restrict__ Bp,
                                                         float* __restrict__ h1) {
  __shared__ __align__(16) unsigned ldsB[B_DWORDS];

  // ---- wave 0: TDM async copy of packed W1 (global -> LDS), 1-D tile of 16384 dwords
  if (threadIdx.x < 32) {
    unsigned long long ga = (unsigned long long)Bp;
    unsigned lds_off = (unsigned)(uintptr_t)&ldsB[0];   // addr[31:0] == LDS offset
    v4u g0;
    g0.x = 1u;                                          // count=1, user descriptor
    g0.y = lds_off;                                     // lds_addr
    g0.z = (unsigned)ga;                                // global_addr[31:0]
    g0.w = (unsigned)((ga >> 32) & 0x01FFFFFFu) | 0x80000000u; // ga[56:32] | type=2
    v8i g1;
    g1[0] = 0x00020000;             // workgroup_mask=0, data_size=2 (4 B)
    g1[1] = (int)(16384u << 16);    // tensor_dim0[15:0] in bits 63:48
    g1[2] = 0;                      // tensor_dim0 hi, tensor_dim1=0
    g1[3] = (int)(16384u << 16);    // tile_dim0 = 16384 (bits 127:112)
    g1[4] = 0;                      // tile_dim1=0, tile_dim2=0
    g1[5] = 16384;                  // tensor_dim0_stride low
    g1[6] = 0;
    g1[7] = 0;
    v4i z4 = {0, 0, 0, 0};
    v8i z8 = {0, 0, 0, 0, 0, 0, 0, 0};
    __builtin_amdgcn_tensor_load_to_lds(g0, g1, z4, z4, z8, 0);
    __builtin_amdgcn_s_wait_tensorcnt(0);
  }
  __syncthreads();

  const int lane = threadIdx.x & 31;
  const int gw = blockIdx.x * 8 + (threadIdx.x >> 5);   // global wave id
  const int row0 = gw * 16;
  if (row0 >= N_NODES) return;                          // tail waves exit after barrier

  const int lm = lane & 15;            // row-within-tile (A) / col (B,C)
  const int lh = lane >> 4;            // k-half selector (A,B) / row-half (C)
  const float* xrow = x + (size_t)(row0 + lm) * IN_DIM;

  union Acc { v8f v; float f[8]; };
  Acc acc[8] = {};                     // 8 N-tiles of 16 cols

  for (int t = 0; t < 8; ++t) {        // K tiles of 32
    const int kA = t * 32 + lh * 8;
    union { v16bf v; unsigned u[8]; } A;
    const float4 f0 = *(const float4*)(xrow + kA + 0);
    const float4 f1 = *(const float4*)(xrow + kA + 4);
    const float4 f2 = *(const float4*)(xrow + kA + 16);
    const float4 f3 = *(const float4*)(xrow + kA + 20);
    A.u[0] = pk2(f0.x, f0.y); A.u[1] = pk2(f0.z, f0.w);
    A.u[2] = pk2(f1.x, f1.y); A.u[3] = pk2(f1.z, f1.w);
    A.u[4] = pk2(f2.x, f2.y); A.u[5] = pk2(f2.z, f2.w);
    A.u[6] = pk2(f3.x, f3.y); A.u[7] = pk2(f3.z, f3.w);
#pragma unroll
    for (int j = 0; j < 8; ++j) {
      union { v16bf v; uint4 q[2]; } B;
      const uint4* bp = (const uint4*)&ldsB[(((t * 8 + j) * 32 + lane) * 8)];
      B.q[0] = bp[0];
      B.q[1] = bp[1];
      acc[j].v = __builtin_amdgcn_wmma_f32_16x16x32_bf16(
          /*neg_a=*/false, A.v, /*neg_b=*/false, B.v,
          /*c_mod=*/(short)0, acc[j].v, /*reuse_a=*/false, /*reuse_b=*/false);
    }
  }
  // C/D layout: VGPR r -> M = r + 8*lh ; N = lm
#pragma unroll
  for (int j = 0; j < 8; ++j)
#pragma unroll
    for (int r = 0; r < 8; ++r)
      h1[(size_t)(row0 + lh * 8 + r) * HID + j * 16 + lm] = acc[j].f[r];
}

// ---------------- per-node attention logits, layer 1 ----------------
__global__ __launch_bounds__(256) void alpha1_kernel(const float* __restrict__ h1,
                                                     const float* __restrict__ att_s,
                                                     const float* __restrict__ att_d,
                                                     float* __restrict__ aS,
                                                     float* __restrict__ aD) {
  int i = blockIdx.x * 256 + threadIdx.x;           // n*HEADS + h
  if (i >= N_NODES * HEADS) return;
  int n = i >> 2, h = i & 3;
  const float* hp = h1 + (size_t)n * HID + h * C1;
  const float* sp = att_s + h * C1;
  const float* dp = att_d + h * C1;
  float s = 0.f, d = 0.f;
#pragma unroll
  for (int c = 0; c < C1; ++c) { float v = hp[c]; s += v * sp[c]; d += v * dp[c]; }
  aS[i] = s; aD[i] = d;
}

// ---------------- init accumulators ----------------
__global__ __launch_bounds__(256) void init_kernel(float* aggr1, float* emax1, float* denom1,
                                                   float* emax2, float* denom2, float* out2) {
  int i = blockIdx.x * 256 + threadIdx.x;
  if (i < N_NODES * HID) aggr1[i] = 0.f;
  if (i < N_NODES * HEADS) { emax1[i] = -INFINITY; denom1[i] = 0.f; }
  if (i < N_NODES) { emax2[i] = -INFINITY; denom2[i] = 0.f; out2[2*i] = 0.f; out2[2*i+1] = 0.f; }
}

// ---------------- edge pass 1: segment max ----------------
__global__ __launch_bounds__(256) void emax1_kernel(const long long* __restrict__ ei,
                                                    const float* __restrict__ aS,
                                                    const float* __restrict__ aD,
                                                    float* __restrict__ emax) {
  int e = blockIdx.x * 256 + threadIdx.x;
  if (e >= ETOT) return;
  int s, d; edge_sd(ei, e, s, d);
#pragma unroll
  for (int h = 0; h < HEADS; ++h) {
    float v = aS[s * HEADS + h] + aD[d * HEADS + h];
    v = v > 0.f ? v : NEG_SLOPE * v;
    atomicMax(emax + d * HEADS + h, v);
  }
}

// ---------------- edge pass 2: denominator ----------------
__global__ __launch_bounds__(256) void denom1_kernel(const long long* __restrict__ ei,
                                                     const float* __restrict__ aS,
                                                     const float* __restrict__ aD,
                                                     const float* __restrict__ emax,
                                                     float* __restrict__ denom) {
  int e = blockIdx.x * 256 + threadIdx.x;
  if (e >= ETOT) return;
  int s, d; edge_sd(ei, e, s, d);
#pragma unroll
  for (int h = 0; h < HEADS; ++h) {
    float v = aS[s * HEADS + h] + aD[d * HEADS + h];
    v = v > 0.f ? v : NEG_SLOPE * v;
    atomicAdd(denom + d * HEADS + h, __expf(v - emax[d * HEADS + h]));
  }
}

// ---------------- edge pass 3: weighted aggregation (dominant gather) ----------------
__global__ __launch_bounds__(256) void aggr1_kernel(const long long* __restrict__ ei,
                                                    const float* __restrict__ aS,
                                                    const float* __restrict__ aD,
                                                    const float* __restrict__ emax,
                                                    const float* __restrict__ denom,
                                                    const float* __restrict__ h1,
                                                    float* __restrict__ out) {
  int i = blockIdx.x * 256 + threadIdx.x;           // (edge, head)
  if (i >= ETOT * HEADS) return;
  int e = i >> 2, h = i & 3;
  int s, d; edge_sd(ei, e, s, d);
  float v = aS[s * HEADS + h] + aD[d * HEADS + h];
  v = v > 0.f ? v : NEG_SLOPE * v;
  float alpha = __expf(v - emax[d * HEADS + h]) / denom[d * HEADS + h];
  const float4* hp = (const float4*)(h1 + (size_t)s * HID + h * C1);
  __builtin_prefetch(hp, 0, 0);                     // global_prefetch_b8
  float* op = out + (size_t)d * HID + h * C1;
#pragma unroll
  for (int c4 = 0; c4 < 8; ++c4) {
    float4 hv = hp[c4];
    atomicAdd(op + c4 * 4 + 0, hv.x * alpha);
    atomicAdd(op + c4 * 4 + 1, hv.y * alpha);
    atomicAdd(op + c4 * 4 + 2, hv.z * alpha);
    atomicAdd(op + c4 * 4 + 3, hv.w * alpha);
  }
}

// ---------------- bias + ELU (in place on aggr1 -> h1f) ----------------
__global__ __launch_bounds__(256) void fin1_kernel(float* __restrict__ aggr1,
                                                   const float* __restrict__ b1) {
  int i = blockIdx.x * 256 + threadIdx.x;
  if (i >= N_NODES * HID) return;
  float v = aggr1[i] + b1[i & (HID - 1)];
  aggr1[i] = v > 0.f ? v : (__expf(v) - 1.0f);
}

// ---------------- GEMM2 (N x 128 @ 128 x 2) + layer-2 logits: one wave32 per node ----
__global__ __launch_bounds__(256) void gemm2_kernel(const float* __restrict__ h1f,
                                                    const float* __restrict__ W2,
                                                    const float* __restrict__ att_s2,
                                                    const float* __restrict__ att_d2,
                                                    float* __restrict__ g,
                                                    float* __restrict__ aS2,
                                                    float* __restrict__ aD2) {
  int wave = (blockIdx.x * 256 + threadIdx.x) >> 5;
  int lane = threadIdx.x & 31;
  if (wave >= N_NODES) return;
  const float4 hv = *(const float4*)(h1f + (size_t)wave * HID + lane * 4);
  float a0 = 0.f, a1 = 0.f;
#pragma unroll
  for (int i = 0; i < 4; ++i) {
    float hx = (i == 0) ? hv.x : (i == 1) ? hv.y : (i == 2) ? hv.z : hv.w;
    a0 += hx * W2[(lane * 4 + i) * 2 + 0];
    a1 += hx * W2[(lane * 4 + i) * 2 + 1];
  }
#pragma unroll
  for (int off = 16; off > 0; off >>= 1) {
    a0 += __shfl_xor(a0, off, 32);
    a1 += __shfl_xor(a1, off, 32);
  }
  if (lane == 0) {
    g[wave * 2 + 0] = a0;
    g[wave * 2 + 1] = a1;
    aS2[wave] = a0 * att_s2[0] + a1 * att_s2[1];
    aD2[wave] = a0 * att_d2[0] + a1 * att_d2[1];
  }
}

// ---------------- layer-2 edge passes (H=1, C=2) ----------------
__global__ __launch_bounds__(256) void emax2_kernel(const long long* __restrict__ ei,
                                                    const float* __restrict__ aS,
                                                    const float* __restrict__ aD,
                                                    float* __restrict__ emax) {
  int e = blockIdx.x * 256 + threadIdx.x;
  if (e >= ETOT) return;
  int s, d; edge_sd(ei, e, s, d);
  float v = aS[s] + aD[d];
  v = v > 0.f ? v : NEG_SLOPE * v;
  atomicMax(emax + d, v);
}

__global__ __launch_bounds__(256) void denom2_kernel(const long long* __restrict__ ei,
                                                     const float* __restrict__ aS,
                                                     const float* __restrict__ aD,
                                                     const float* __restrict__ emax,
                                                     float* __restrict__ denom) {
  int e = blockIdx.x * 256 + threadIdx.x;
  if (e >= ETOT) return;
  int s, d; edge_sd(ei, e, s, d);
  float v = aS[s] + aD[d];
  v = v > 0.f ? v : NEG_SLOPE * v;
  atomicAdd(denom + d, __expf(v - emax[d]));
}

__global__ __launch_bounds__(256) void aggr2_kernel(const long long* __restrict__ ei,
                                                    const float* __restrict__ aS,
                                                    const float* __restrict__ aD,
                                                    const float* __restrict__ emax,
                                                    const float* __restrict__ denom,
                                                    const float* __restrict__ g,
                                                    float* __restrict__ out2) {
  int e = blockIdx.x * 256 + threadIdx.x;
  if (e >= ETOT) return;
  int s, d; edge_sd(ei, e, s, d);
  float v = aS[s] + aD[d];
  v = v > 0.f ? v : NEG_SLOPE * v;
  float alpha = __expf(v - emax[d]) / denom[d];
  atomicAdd(out2 + d * 2 + 0, g[s * 2 + 0] * alpha);
  atomicAdd(out2 + d * 2 + 1, g[s * 2 + 1] * alpha);
}

__global__ __launch_bounds__(256) void final_kernel(const float* __restrict__ out2,
                                                    const float* __restrict__ b2,
                                                    float* __restrict__ out) {
  int n = blockIdx.x * 256 + threadIdx.x;
  if (n >= N_NODES) return;
  out[n * 2 + 0] = out2[n * 2 + 0] + b2[0];   // mean over single head == identity
  out[n * 2 + 1] = out2[n * 2 + 1] + b2[1];
}

// ---------------- host launcher ----------------
extern "C" void kernel_launch(void* const* d_in, const int* in_sizes, int n_in,
                              void* d_out, int out_size, void* d_ws, size_t ws_size,
                              hipStream_t stream) {
  const float*     x        = (const float*)d_in[0];
  const long long* ei       = (const long long*)d_in[1];  // int64 per reference
  const float*     W1       = (const float*)d_in[2];
  const float*     att_src1 = (const float*)d_in[3];
  const float*     att_dst1 = (const float*)d_in[4];
  const float*     b1       = (const float*)d_in[5];
  const float*     W2       = (const float*)d_in[6];
  const float*     att_src2 = (const float*)d_in[7];
  const float*     att_dst2 = (const float*)d_in[8];
  const float*     b2       = (const float*)d_in[9];
  float*           out      = (float*)d_out;

  char* ws = (char*)d_ws;
  size_t off = 0;
  auto alloc = [&](size_t bytes) -> char* {
    char* p = ws + off;
    off += (bytes + 255) & ~(size_t)255;
    return p;
  };
  unsigned* Bp    = (unsigned*)alloc(B_DWORDS * sizeof(unsigned)); // 64 KB
  float* h1raw    = (float*)alloc((size_t)N_NODES * HID * sizeof(float));
  float* aggr1    = (float*)alloc((size_t)N_NODES * HID * sizeof(float));
  float* aS1      = (float*)alloc((size_t)N_NODES * HEADS * sizeof(float));
  float* aD1      = (float*)alloc((size_t)N_NODES * HEADS * sizeof(float));
  float* emax1    = (float*)alloc((size_t)N_NODES * HEADS * sizeof(float));
  float* denom1   = (float*)alloc((size_t)N_NODES * HEADS * sizeof(float));
  float* g        = (float*)alloc((size_t)N_NODES * OUT_C * sizeof(float));
  float* aS2      = (float*)alloc((size_t)N_NODES * sizeof(float));
  float* aD2      = (float*)alloc((size_t)N_NODES * sizeof(float));
  float* emax2    = (float*)alloc((size_t)N_NODES * sizeof(float));
  float* denom2   = (float*)alloc((size_t)N_NODES * sizeof(float));
  float* out2     = (float*)alloc((size_t)N_NODES * OUT_C * sizeof(float));

  const int TB = 256;
  auto blk = [](long long n, int tb) { return (unsigned)((n + tb - 1) / tb); };

  packW1_kernel<<<blk(16384, TB), TB, 0, stream>>>(W1, Bp);
  init_kernel<<<blk((long long)N_NODES * HID, TB), TB, 0, stream>>>(
      aggr1, emax1, denom1, emax2, denom2, out2);
  // 8 waves/block, each wave = 16 rows -> ceil(6250 waves / 8)
  gemm1_wmma_kernel<<<(6250 + 7) / 8, 256, 0, stream>>>(x, Bp, h1raw);
  alpha1_kernel<<<blk((long long)N_NODES * HEADS, TB), TB, 0, stream>>>(
      h1raw, att_src1, att_dst1, aS1, aD1);
  emax1_kernel<<<blk(ETOT, TB), TB, 0, stream>>>(ei, aS1, aD1, emax1);
  denom1_kernel<<<blk(ETOT, TB), TB, 0, stream>>>(ei, aS1, aD1, emax1, denom1);
  aggr1_kernel<<<blk((long long)ETOT * HEADS, TB), TB, 0, stream>>>(
      ei, aS1, aD1, emax1, denom1, h1raw, aggr1);
  fin1_kernel<<<blk((long long)N_NODES * HID, TB), TB, 0, stream>>>(aggr1, b1);
  gemm2_kernel<<<blk((long long)N_NODES * 32, TB), TB, 0, stream>>>(
      aggr1, W2, att_src2, att_dst2, g, aS2, aD2);
  emax2_kernel<<<blk(ETOT, TB), TB, 0, stream>>>(ei, aS2, aD2, emax2);
  denom2_kernel<<<blk(ETOT, TB), TB, 0, stream>>>(ei, aS2, aD2, emax2, denom2);
  aggr2_kernel<<<blk(ETOT, TB), TB, 0, stream>>>(ei, aS2, aD2, emax2, denom2, g, out2);
  final_kernel<<<blk(N_NODES, TB), TB, 0, stream>>>(out2, b2, out);
}
